// GeneratorIto_70351564308751
// MI455X (gfx1250) — matrix operand order repaired and checked by reading, not compiled, not run on hardware
//
#include <hip/hip_runtime.h>
#include <math.h>

#define T_STEPS 256
#define NPART   65536
#define HID     64
#define FWD_RATE 0.03f

typedef __attribute__((ext_vector_type(2)))  float     v2f;
typedef __attribute__((ext_vector_type(8)))  float     v8f;
typedef __attribute__((ext_vector_type(16))) _Float16  v16h;
typedef __attribute__((ext_vector_type(2)))  __fp16    fp16x2;

#if defined(__has_builtin)
#if __has_builtin(__builtin_amdgcn_tanhf)
#define TANH(x) __builtin_amdgcn_tanhf(x)
#elif __has_builtin(__builtin_amdgcn_tanh_f32)
#define TANH(x) __builtin_amdgcn_tanh_f32(x)
#else
#define TANH(x) tanhf(x)
#endif
#else
#define TANH(x) tanhf(x)
#endif

union HPack { unsigned u[8]; v16h v; };

// pack two f32 -> packed f16 pair (v_cvt_pkrtz_f16_f32)
__device__ __forceinline__ unsigned pack2(float a, float b) {
  fp16x2 p = __builtin_amdgcn_cvt_pkrtz(a, b);
  return __builtin_bit_cast(unsigned, p);
}
// swap 16-lane halves (identity position) via v_permlanex16_b32
__device__ __forceinline__ unsigned xhalf_u(unsigned v) {
  return __builtin_amdgcn_permlanex16(v, v, 0x76543210u, 0xfedcba98u, false, false);
}
__device__ __forceinline__ float xhalf_f(float v) {
  unsigned iv = __builtin_bit_cast(unsigned, v);
  unsigned rv = __builtin_amdgcn_permlanex16(iv, iv, 0x76543210u, 0xfedcba98u, false, false);
  return __builtin_bit_cast(float, rv);
}

__global__ __launch_bounds__(256) void sde_paths_kernel(
    const float* __restrict__ t,  const float* __restrict__ xy0,
    const float* __restrict__ dW, const float* __restrict__ W1,
    const float* __restrict__ b1, const float* __restrict__ W2,
    const float* __restrict__ b2, float* __restrict__ out)
{
  const int  lane = threadIdx.x & 31;
  const int  m    = lane & 15;          // particle slot within wave / matrix row index
  const bool hi   = (lane & 16) != 0;   // upper 16-lane half
  const int  gwave = blockIdx.x * 8 + (threadIdx.x >> 5);
  const int  p     = gwave * 16 + m;    // global particle id

  const size_t o_xy = 256;
  const size_t o_r  = o_xy + (size_t)T_STEPS * NPART * 2;
  const size_t o_bm = o_r  + (size_t)T_STEPS * NPART;
  // per-lane float2 output region: lower half -> xy path, upper half -> BM path
  const size_t ofs1 = hi ? o_bm : o_xy;

  // ---- Layer-1 A matrix: W1ext^T chunks (16 hidden rows x K=4), f32 A-layout.
  // lane<16 holds K={0:t-coef, 1:x-coef}; lane>=16 holds K={2:y-coef, 3:bias}.
  v2f A1[4];
#pragma unroll
  for (int c = 0; c < 4; ++c) {
    const int hid = 16 * c + m;
    A1[c].x = hi ? W1[2 * HID + hid] : W1[0 * HID + hid];
    A1[c].y = hi ? b1[hid]           : W1[1 * HID + hid];
  }

  // ---- Layer-2 A matrix: W2^T padded to 16 rows (16 x 64), f16 A-layout, 2 K-halves.
  HPack A2[2];
#pragma unroll
  for (int kh = 0; kh < 2; ++kh) {
#pragma unroll
    for (int v = 0; v < 8; ++v) {
      const int k0 = (hi ? 16 : 0) + 2 * v + 32 * kh;
      const float w0 = (m < 5) ? W2[k0 * 5 + m]       : 0.0f;
      const float w1 = (m < 5) ? W2[(k0 + 1) * 5 + m] : 0.0f;
      A2[kh].u[v] = pack2(w0, w1);
    }
  }

  float b2c[5];
#pragma unroll
  for (int k = 0; k < 5; ++k) b2c[k] = b2[k];

  float x = xy0[2 * p], y = xy0[2 * p + 1], r = 0.0f, cum = 0.0f;
  const float dt  = t[1] - t[0];
  const float sdt = sqrtf(dt);
  float ti = t[0];                // running time; dt = 2^-8 -> exact accumulation

  {  // row 0 of each output, all lanes active
    const float2 s0 = hi ? make_float2(0.0f, 0.0f) : make_float2(x, y);
    ((float2*)(out + ofs1))[p] = s0;
    out[o_r + p] = 0.0f;         // both halves write identical value
  }

  // software-pipelined dW load (hide VMEM latency under WMMA/tanh chain)
  float2 dwv = ((const float2*)dW)[p];  // step 0

  const v8f zero8 = {};
  for (int i = 0; i < T_STEPS - 1; ++i) {
    const int inext = (i < T_STEPS - 2) ? (i + 1) : (T_STEPS - 2);  // uniform clamp
    const float2 dwv_n = ((const float2*)dW)[(size_t)inext * NPART + p];

    // B1 = [t; x; y; 1] columns = particles, f32 B-layout (K split across halves)
    v2f B1;
    B1.x = hi ? y    : ti;
    B1.y = hi ? 1.0f : x;

    // h^T chunks: 4x v_wmma_f32_16x16x4_f32
    v8f C1[4];
#pragma unroll
    for (int c = 0; c < 4; ++c)
      C1[c] = __builtin_amdgcn_wmma_f32_16x16x4_f32(false, A1[c], false, B1,
                                                    (short)0, zero8, false, false);

    // tanh + pack adjacent hidden rows into f16 pairs
    unsigned P[4][4];
#pragma unroll
    for (int c = 0; c < 4; ++c)
#pragma unroll
      for (int j = 0; j < 4; ++j)
        P[c][j] = pack2(TANH(C1[c][2 * j]), TANH(C1[c][2 * j + 1]));

    // rearrange C-layout (rows r / r+8 across halves) -> f16 B operand in this code's k order (rows 0-15 lower, 16-31 upper, same as A2)
    HPack B2[2];
#pragma unroll
    for (int kh = 0; kh < 2; ++kh) {
      const int ca = 2 * kh, cb = 2 * kh + 1;
#pragma unroll
      for (int v = 0; v < 4; ++v) {
        const unsigned qu = xhalf_u(P[cb][v]);
        const unsigned ql = xhalf_u(P[ca][v]);
        B2[kh].u[v]     = hi ? qu        : P[ca][v];
        B2[kh].u[4 + v] = hi ? P[cb][v]  : ql;
      }
    }

    // out^T = W2^T @ h^T : two chained v_wmma_f32_16x16x32_f16 over K=64
    v8f D = __builtin_amdgcn_wmma_f32_16x16x32_f16(false, A2[0].v, false, B2[0].v,
                                                   (short)0, zero8, false, false);
    D     = __builtin_amdgcn_wmma_f32_16x16x32_f16(false, A2[1].v, false, B2[1].v,
                                                   (short)0, D,     false, false);

    // D VGPR k, lower lanes: output k of particle=lane; replicate to upper half
    float V[5];
#pragma unroll
    for (int k = 0; k < 5; ++k) {
      const float q = xhalf_f(D[k]);
      V[k] = (hi ? q : D[k]) + b2c[k];
    }

    // Euler-Maruyama update
    const float dws0 = dwv.x * sdt, dws1 = dwv.y * sdt;
    const float Sig  = V[0] * V[0] + V[1] * V[1];
    x += (FWD_RATE - 0.5f * Sig) * dt + V[0] * dws0 + V[1] * dws1;
    y += V[4] * dt + V[2] * dws0 + V[3] * dws1;
    r += 0.5f * Sig * dt;
    cum += dws0;

    // all-lane stores: lower half -> xy, upper half -> BM; r duplicated (same value)
    {
      const size_t row = (size_t)(i + 1) * NPART + p;
      const float2 s1 = hi ? make_float2(cum, cum) : make_float2(x, y);
      ((float2*)(out + ofs1))[row] = s1;
      out[o_r + row] = r;
    }

    dwv = dwv_n;
    ti += dt;
  }
}

__global__ void copy_t_kernel(const float* __restrict__ t, float* __restrict__ out) {
  out[threadIdx.x] = t[threadIdx.x];
}

extern "C" void kernel_launch(void* const* d_in, const int* in_sizes, int n_in,
                              void* d_out, int out_size, void* d_ws, size_t ws_size,
                              hipStream_t stream) {
  const float* t   = (const float*)d_in[0];
  const float* xy0 = (const float*)d_in[1];
  const float* dW  = (const float*)d_in[2];
  const float* W1  = (const float*)d_in[3];
  const float* b1  = (const float*)d_in[4];
  const float* W2  = (const float*)d_in[5];
  const float* b2  = (const float*)d_in[6];
  float* out = (float*)d_out;

  copy_t_kernel<<<1, 256, 0, stream>>>(t, out);
  // 16 particles per wave, 8 waves per block -> 65536 / 128 = 512 blocks
  sde_paths_kernel<<<NPART / 128, 256, 0, stream>>>(t, xy0, dW, W1, b1, W2, b2, out);
}
